// RecGATblock_37838661877769
// MI455X (gfx1250) — compile-verified, matrix operand order checked by reading, not run on hardware
//
#include <hip/hip_runtime.h>
#include <hip/hip_bf16.h>

typedef __attribute__((ext_vector_type(16))) __bf16 v16bf;
typedef __attribute__((ext_vector_type(8)))  float  v8f;

#define N_NODES   50000
#define N_EDGES   800000
#define IN_FEATS  32
#define OUT_FEATS 32
#define NUM_HEADS 4
#define HF        128   // H*F
#define G3        96    // 3*F

// ---------- WMMA lane/K mapping helpers (16-bit A/B, 16x16x32) ----------
// A (16x32, MxK): lane L holds row M=L&15. Half-wave K split:
//   lanes 0-15: elements 0..7 -> K=0..7,  elements 8..15 -> K=16..23
//   lanes16-31: elements 0..7 -> K=8..15, elements 8..15 -> K=24..31
// B (32x16, KxN): lane L holds column N=L&15, same K striping.
// C/D (16x16 f32): VGPR r holds M = r + (lane<16?0:8), N = lane&15.

__device__ __forceinline__ void pack_a(v16bf& a, float4 f0, float4 f1, float4 f2, float4 f3) {
    a[0]=(__bf16)f0.x;  a[1]=(__bf16)f0.y;  a[2]=(__bf16)f0.z;  a[3]=(__bf16)f0.w;
    a[4]=(__bf16)f1.x;  a[5]=(__bf16)f1.y;  a[6]=(__bf16)f1.z;  a[7]=(__bf16)f1.w;
    a[8]=(__bf16)f2.x;  a[9]=(__bf16)f2.y;  a[10]=(__bf16)f2.z; a[11]=(__bf16)f2.w;
    a[12]=(__bf16)f3.x; a[13]=(__bf16)f3.y; a[14]=(__bf16)f3.z; a[15]=(__bf16)f3.w;
}

// ---------- ordered-uint encoding for float atomic max ----------
__device__ __forceinline__ unsigned f2key(float x) {
    unsigned u = __float_as_uint(x);
    return (u & 0x80000000u) ? ~u : (u | 0x80000000u);
}
__device__ __forceinline__ float key2f(unsigned k) {
    unsigned u = (k & 0x80000000u) ? (k ^ 0x80000000u) : ~k;
    return __uint_as_float(u);
}

// ============ Kernel 1: feat = h @ W_gat  (N,32)x(32,128) via WMMA bf16 ============
// grid = N/16 row tiles, 8 waves/block = 8 column tiles of 16.
__global__ __launch_bounds__(256) void k_feat(const float* __restrict__ h,
                                              const float* __restrict__ Wg,
                                              float* __restrict__ feat) {
    const int lane = threadIdx.x & 31;
    const int wave = threadIdx.x >> 5;          // column tile 0..7
    const int m0   = blockIdx.x * 16;
    const int row  = m0 + (lane & 15);
    const int kb   = (lane < 16) ? 0 : 8;

    // A tile: row of h, 32 K values, two contiguous 8-float chunks per lane
    const float4* hp = (const float4*)(h + row * IN_FEATS);
    v16bf a;
    pack_a(a, hp[kb/4], hp[kb/4 + 1], hp[(kb+16)/4], hp[(kb+16)/4 + 1]);

    // B tile: column of W_gat (K-major, stride HF)
    const int col = wave * 16 + (lane & 15);
    v16bf b;
#pragma unroll
    for (int i = 0; i < 16; ++i) {
        int k = kb + ((i < 8) ? i : i + 8);
        b[i] = (__bf16)Wg[k * HF + col];
    }

    v8f c = {};
    c = __builtin_amdgcn_wmma_f32_16x16x32_bf16(false, a, false, b, (short)0, c, false, false);

    const int rbase = m0 + ((lane < 16) ? 0 : 8);
#pragma unroll
    for (int r = 0; r < 8; ++r)
        feat[(rbase + r) * HF + col] = c[r];
}

// ============ Kernel 2: el/er head-wise dot products ============
__global__ void k_elr(const float* __restrict__ feat, const float* __restrict__ al,
                      const float* __restrict__ ar, float* __restrict__ el,
                      float* __restrict__ er) {
    int t = blockIdx.x * blockDim.x + threadIdx.x;
    if (t >= N_NODES * NUM_HEADS) return;
    int n = t >> 2, hd = t & 3;
    const float* fp  = feat + n * HF + hd * OUT_FEATS;
    const float* alp = al + hd * OUT_FEATS;
    const float* arp = ar + hd * OUT_FEATS;
    float sl = 0.f, sr = 0.f;
#pragma unroll 8
    for (int f = 0; f < OUT_FEATS; ++f) {
        float v = fp[f];
        sl += v * alp[f];
        sr += v * arp[f];
    }
    el[t] = sl; er[t] = sr;
}

// ============ Kernel 3: init accumulators ============
__global__ void k_init(float* __restrict__ agg, float* __restrict__ denom,
                       unsigned* __restrict__ emax) {
    int t = blockIdx.x * blockDim.x + threadIdx.x;
    if (t < N_NODES * HF) agg[t] = 0.f;
    if (t < N_NODES * NUM_HEADS) { denom[t] = 0.f; emax[t] = 0u; }
}

// ============ Kernel 4: edge logits + segment max (one thread per edge, 4 heads) ============
__global__ void k_edge_logit(const int* __restrict__ src, const int* __restrict__ dst,
                             const float* __restrict__ el, const float* __restrict__ er,
                             float* __restrict__ elog, unsigned* __restrict__ emax) {
    int e = blockIdx.x * blockDim.x + threadIdx.x;
    if (e >= N_EDGES) return;
    int s = src[e], d = dst[e];
    float4 l = *(const float4*)(el + s * NUM_HEADS);
    float4 r = *(const float4*)(er + d * NUM_HEADS);
    float v0 = l.x + r.x, v1 = l.y + r.y, v2 = l.z + r.z, v3 = l.w + r.w;
    v0 = (v0 > 0.f) ? v0 : 0.2f * v0;
    v1 = (v1 > 0.f) ? v1 : 0.2f * v1;
    v2 = (v2 > 0.f) ? v2 : 0.2f * v2;
    v3 = (v3 > 0.f) ? v3 : 0.2f * v3;
    *(float4*)(elog + e * NUM_HEADS) = make_float4(v0, v1, v2, v3);
    unsigned* em = emax + d * NUM_HEADS;
    atomicMax(em + 0, f2key(v0));
    atomicMax(em + 1, f2key(v1));
    atomicMax(em + 2, f2key(v2));
    atomicMax(em + 3, f2key(v3));
}

// ============ Kernel 5: exp(e - max) + segment sum (one thread per edge) ============
__global__ void k_edge_exp(const int* __restrict__ dst, float* __restrict__ elog,
                           const unsigned* __restrict__ emax, float* __restrict__ denom) {
    int e = blockIdx.x * blockDim.x + threadIdx.x;
    if (e >= N_EDGES) return;
    int d = dst[e];
    uint4  mk = *(const uint4*)(emax + d * NUM_HEADS);
    float4 v  = *(const float4*)(elog + e * NUM_HEADS);
    float e0 = __expf(v.x - key2f(mk.x));
    float e1 = __expf(v.y - key2f(mk.y));
    float e2 = __expf(v.z - key2f(mk.z));
    float e3 = __expf(v.w - key2f(mk.w));
    *(float4*)(elog + e * NUM_HEADS) = make_float4(e0, e1, e2, e3);
    float* dn = denom + d * NUM_HEADS;
    atomicAdd(dn + 0, e0);
    atomicAdd(dn + 1, e1);
    atomicAdd(dn + 2, e2);
    atomicAdd(dn + 3, e3);
}

// ============ Kernel 6: alpha-weighted message scatter (bandwidth-dominant) ============
// One thread per (edge, 4-feature chunk): float4 read of feat, 4 atomic f32 adds.
// Index/alpha loads amortized 4x vs per-feature threads.
__global__ void k_scatter(const int* __restrict__ src, const int* __restrict__ dst,
                          const float* __restrict__ elog, const float* __restrict__ denom,
                          const float* __restrict__ feat, float* __restrict__ agg) {
    int t = blockIdx.x * blockDim.x + threadIdx.x;   // E * 32 = 25.6M threads, fits int
    if (t >= N_EDGES * (HF / 4)) return;
    int e  = t >> 5;             // edge
    int q  = t & 31;             // 4-feature chunk 0..31
    int c  = q << 2;             // feature base 0..124
    int hd = q >> 3;             // head = c / 32
    int s = src[e], d = dst[e];
    float alpha = elog[e * NUM_HEADS + hd] / denom[d * NUM_HEADS + hd];
    float4 f = *(const float4*)(feat + s * HF + c);
    float* ap = agg + d * HF + c;
    atomicAdd(ap + 0, f.x * alpha);
    atomicAdd(ap + 1, f.y * alpha);
    atomicAdd(ap + 2, f.z * alpha);
    atomicAdd(ap + 3, f.w * alpha);
}

// ============ Kernel 7: GRU gemms gi = agg@W_ihT + b_ih ; gh = h@W_hhT + b_hh ============
// grid = N/16 row tiles, 6 waves/block = 6 column tiles covering 96 gate columns.
__global__ __launch_bounds__(192) void k_gru_gemm(const float* __restrict__ agg,
                                                  const float* __restrict__ h,
                                                  const float* __restrict__ Wih,
                                                  const float* __restrict__ Whh,
                                                  const float* __restrict__ bih,
                                                  const float* __restrict__ bhh,
                                                  float* __restrict__ gi,
                                                  float* __restrict__ gh) {
    const int lane = threadIdx.x & 31;
    const int wave = threadIdx.x >> 5;          // column tile 0..5
    const int m0   = blockIdx.x * 16;
    const int kb   = (lane < 16) ? 0 : 8;
    const int col  = wave * 16 + (lane & 15);   // gate column 0..95
    const int row  = m0 + (lane & 15);
    const int rbase = m0 + ((lane < 16) ? 0 : 8);

    // ---- gi: K = 128 -> 4 WMMA steps, B[k][j] = Wih[j*128 + k] ----
    v8f c = {};
#pragma unroll
    for (int kk = 0; kk < 4; ++kk) {
        const int kbase = kk * 32;
        const float* ap = agg + row * HF + kbase;
        v16bf a, b;
        pack_a(a, *(const float4*)(ap + kb),      *(const float4*)(ap + kb + 4),
                  *(const float4*)(ap + kb + 16), *(const float4*)(ap + kb + 20));
#pragma unroll
        for (int i = 0; i < 16; ++i) {
            int k = kbase + kb + ((i < 8) ? i : i + 8);
            b[i] = (__bf16)Wih[col * HF + k];
        }
        c = __builtin_amdgcn_wmma_f32_16x16x32_bf16(false, a, false, b, (short)0, c, false, false);
    }
    const float bi = bih[col];
#pragma unroll
    for (int r = 0; r < 8; ++r)
        gi[(rbase + r) * G3 + col] = c[r] + bi;

    // ---- gh: K = 32 -> 1 WMMA step, B[k][j] = Whh[j*32 + k] ----
    {
        const float4* hp = (const float4*)(h + row * IN_FEATS);
        v16bf a, b;
        pack_a(a, hp[kb/4], hp[kb/4 + 1], hp[(kb+16)/4], hp[(kb+16)/4 + 1]);
#pragma unroll
        for (int i = 0; i < 16; ++i) {
            int k = kb + ((i < 8) ? i : i + 8);
            b[i] = (__bf16)Whh[col * IN_FEATS + k];
        }
        v8f c2 = {};
        c2 = __builtin_amdgcn_wmma_f32_16x16x32_bf16(false, a, false, b, (short)0, c2, false, false);
        const float bh = bhh[col];
#pragma unroll
        for (int r = 0; r < 8; ++r)
            gh[(rbase + r) * G3 + col] = c2[r] + bh;
    }
}

// ============ Kernel 8: fused GRU gates + ELU ============
__global__ void k_gates(const float* __restrict__ gi, const float* __restrict__ gh,
                        const float* __restrict__ h, float* __restrict__ out) {
    int t = blockIdx.x * blockDim.x + threadIdx.x;
    if (t >= N_NODES * OUT_FEATS) return;
    int n = t >> 5, f = t & 31;
    float ir = gi[n * G3 + f],      iz = gi[n * G3 + 32 + f], in_ = gi[n * G3 + 64 + f];
    float hr = gh[n * G3 + f],      hz = gh[n * G3 + 32 + f], hn  = gh[n * G3 + 64 + f];
    float r  = 1.f / (1.f + __expf(-(ir + hr)));
    float z  = 1.f / (1.f + __expf(-(iz + hz)));
    float nn = tanhf(in_ + r * hn);
    float hv = h[t];
    float hnew = (1.f - z) * nn + z * hv;
    out[t] = (hnew > 0.f) ? hnew : (__expf(hnew) - 1.f);   // ELU
}

extern "C" void kernel_launch(void* const* d_in, const int* in_sizes, int n_in,
                              void* d_out, int out_size, void* d_ws, size_t ws_size,
                              hipStream_t stream) {
    const float* h     = (const float*)d_in[0];
    const float* Wgat  = (const float*)d_in[1];
    const float* attnl = (const float*)d_in[2];
    const float* attnr = (const float*)d_in[3];
    const float* Wih   = (const float*)d_in[4];
    const float* Whh   = (const float*)d_in[5];
    const float* bih   = (const float*)d_in[6];
    const float* bhh   = (const float*)d_in[7];
    const int*   src   = (const int*)d_in[8];
    const int*   dst   = (const int*)d_in[9];
    float* out = (float*)d_out;

    // ---- workspace partition (256-byte aligned slabs) ----
    char* ws = (char*)d_ws;
    size_t off = 0;
    auto slab = [&](size_t bytes) -> char* {
        char* p = ws + off;
        off = (off + bytes + 255) & ~(size_t)255;
        return p;
    };
    float*    feat  = (float*)   slab((size_t)N_NODES * HF * 4);
    float*    el    = (float*)   slab((size_t)N_NODES * NUM_HEADS * 4);
    float*    er    = (float*)   slab((size_t)N_NODES * NUM_HEADS * 4);
    float*    elog  = (float*)   slab((size_t)N_EDGES * NUM_HEADS * 4);
    unsigned* emax  = (unsigned*)slab((size_t)N_NODES * NUM_HEADS * 4);
    float*    denom = (float*)   slab((size_t)N_NODES * NUM_HEADS * 4);
    float*    agg   = (float*)   slab((size_t)N_NODES * HF * 4);
    float*    gi    = (float*)   slab((size_t)N_NODES * G3 * 4);
    float*    gh    = (float*)   slab((size_t)N_NODES * G3 * 4);
    (void)ws_size; (void)in_sizes; (void)n_in; (void)out_size;

    const int ROW_TILES = N_NODES / 16;   // 3125, exact

    k_feat<<<ROW_TILES, 256, 0, stream>>>(h, Wgat, feat);
    k_init<<<(N_NODES * HF + 255) / 256, 256, 0, stream>>>(agg, denom, emax);
    k_elr<<<(N_NODES * NUM_HEADS + 255) / 256, 256, 0, stream>>>(feat, attnl, attnr, el, er);
    k_edge_logit<<<(N_EDGES + 255) / 256, 256, 0, stream>>>(src, dst, el, er, elog, emax);
    k_edge_exp<<<(N_EDGES + 255) / 256, 256, 0, stream>>>(dst, elog, emax, denom);
    k_scatter<<<(N_EDGES * (HF / 4) + 255) / 256, 256, 0, stream>>>(src, dst, elog, denom, feat, agg);
    k_gru_gemm<<<ROW_TILES, 192, 0, stream>>>(agg, h, Wih, Whh, bih, bhh, gi, gh);
    k_gates<<<(N_NODES * OUT_FEATS + 255) / 256, 256, 0, stream>>>(gi, gh, h, out);
}
